// MLFF_dmirror_82626580840900
// MI455X (gfx1250) — compile-verified, matrix operand order checked by reading, not run on hardware
//
#include <hip/hip_runtime.h>
#include <hip/hip_bf16.h>
#include <math.h>

// ---------------------------------------------------------------------------
// Problem constants (from the reference)
// ---------------------------------------------------------------------------
#define Bb   4
#define Nn   256
#define Ff   64
#define Hh1  128
#define Hh2  64

typedef __attribute__((ext_vector_type(16))) _Float16 v16h;
typedef __attribute__((ext_vector_type(8)))  float    v8f;
typedef __attribute__((ext_vector_type(4)))  float    v4f;

// ---------------------------------------------------------------------------
// WMMA fragment helpers (CDNA5 ISA 7.12.2 layouts, wave32)
//
// 16-bit A matrix 16x32 (MxK):
//   lane l: M = l & 15 ; kbase = (l < 16) ? 0 : 8
//   half slot hi in 0..15:  K = kbase + (hi < 8 ? hi : hi + 8)
// B matrix 32x16 (KxN) mirrors A with N = l & 15.
// f32 C/D 16x16: lane l, vgpr r:  M = r + 8*(l>>4), N = l & 15.
// ---------------------------------------------------------------------------
__device__ __forceinline__ int frag_k(int hi, int lane) {
    int kbase = (lane & 16) ? 8 : 0;
    return kbase + (hi < 8 ? hi : hi + 8);
}

// A fragment (16x32 slice starting at column k0) from an LDS fp32 matrix
__device__ __forceinline__ v16h load_a_frag(const float* s, int ldm, int k0, int lane) {
    int m = lane & 15;
    v16h a;
#pragma unroll
    for (int hi = 0; hi < 16; ++hi)
        a[hi] = (_Float16)s[m * ldm + k0 + frag_k(hi, lane)];
    return a;
}

// B fragment (32x16 slice), element (k, n) = W[(k0+k)*ldw + n0+n]  (row-major W)
__device__ __forceinline__ v16h load_b_frag(const float* __restrict__ W, int ldw,
                                            int k0, int n0, int lane) {
    int n = lane & 15;
    v16h b;
#pragma unroll
    for (int hi = 0; hi < 16; ++hi)
        b[hi] = (_Float16)W[(k0 + frag_k(hi, lane)) * ldw + n0 + n];
    return b;
}

// B fragment of W^T: element (k, n) = W[(n0+n)*ldw + k0+k]
__device__ __forceinline__ v16h load_bt_frag(const float* __restrict__ W, int ldw,
                                             int k0, int n0, int lane) {
    int n = lane & 15;
    v16h b;
#pragma unroll
    for (int hi = 0; hi < 16; ++hi)
        b[hi] = (_Float16)W[(n0 + n) * ldw + k0 + frag_k(hi, lane)];
    return b;
}

__device__ __forceinline__ v8f wmma16(v16h a, v16h b, v8f c) {
    // D = A*B + C ; f16 inputs, f32 accumulate (v_wmma_f32_16x16x32_f16)
    return __builtin_amdgcn_wmma_f32_16x16x32_f16(
        /*neg_a=*/false, a, /*neg_b=*/false, b,
        /*c_mod=*/(short)0, c, /*reuse_a=*/false, /*reuse_b=*/false);
}

// Fast activation pair using raw HW transcendentals:
//   e = 2^(z*log2e)          (v_exp_f32)
//   softplus = log2(1+e)*ln2 (v_log_f32 + mul)
//   sigmoid  = e * rcp(1+e)  (v_rcp_f32 + mul)
// Large-z select avoids inf/inf; large-negative z falls out naturally (e->0).
__device__ __forceinline__ void act_pair(float z, float& sp, float& sg) {
    float e  = __builtin_amdgcn_exp2f(z * 1.44269504088896f);
    float d  = 1.f + e;
    float r  = __builtin_amdgcn_rcpf(d);
    float lg = __builtin_amdgcn_logf(d) * 0.69314718055995f;
    bool big = (z > 20.f);
    sp = big ? z   : lg;
    sg = big ? 1.f : e * r;
}

// ---------------------------------------------------------------------------
// Kernel A: per-atom MLP forward + analytic input gradient.
// One wave32 per 16-atom tile; 64 WMMAs per wave (fwd L1, fwd L2, bwd L2, bwd L1).
// Writes g[B*N*F] and per-tile energy partials to workspace.
// ---------------------------------------------------------------------------
__global__ __launch_bounds__(32) void mlff_mlp_grad_kernel(
    const float* __restrict__ image,
    const float* __restrict__ W1, const float* __restrict__ b1,
    const float* __restrict__ W2, const float* __restrict__ b2,
    const float* __restrict__ W3, const float* __restrict__ b3,
    float* __restrict__ g, float* __restrict__ partE)
{
    __shared__ float sX  [16 * Ff];    // input tile
    __shared__ float sA1 [16 * Hh1];   // softplus(z1)
    __shared__ float sS1 [16 * Hh1];   // sigmoid(z1)
    __shared__ float sA2 [16 * Hh2];   // softplus(z2)
    __shared__ float sS2 [16 * Hh2];   // sigmoid(z2)
    __shared__ float sDZ2[16 * Hh2];   // W3 * sigmoid(z2)
    __shared__ float sDZ1[16 * Hh1];   // dE/dz1
    __shared__ float sE  [16];

    const int lane = threadIdx.x;
    const int tile = blockIdx.x;            // 0..63 ; 16 tiles per batch
    const int b    = tile >> 4;
    const int m0   = (tile & 15) << 4;      // first atom of tile
    const int mrow = ((lane >> 4) << 3);    // C-layout M base for this lane half
    const int ncol = lane & 15;

    // ---- stage input tile (contiguous 16*64 floats) ----
    const float* xin = image + (size_t)(b * Nn + m0) * Ff;
    for (int i = lane; i < 16 * Ff; i += 32) sX[i] = xin[i];
    __syncthreads();

    // ---- layer 1: z1 = X @ W1 + b1  (K = 64 -> 2 WMMAs per 16-col tile) ----
    v16h ax0 = load_a_frag(sX, Ff, 0,  lane);
    v16h ax1 = load_a_frag(sX, Ff, 32, lane);
#pragma unroll
    for (int n = 0; n < Hh1 / 16; ++n) {
        v8f c = {};
        c = wmma16(ax0, load_b_frag(W1, Hh1, 0,  n * 16, lane), c);
        c = wmma16(ax1, load_b_frag(W1, Hh1, 32, n * 16, lane), c);
        const int col  = n * 16 + ncol;
        const float bv = b1[col];
#pragma unroll
        for (int r = 0; r < 8; ++r) {
            int m   = mrow + r;
            float z = c[r] + bv;
            float sp, sg;
            act_pair(z, sp, sg);
            sA1[m * Hh1 + col] = sp;
            sS1[m * Hh1 + col] = sg;
        }
    }
    __syncthreads();

    // ---- layer 2: z2 = a1 @ W2 + b2  (K = 128 -> 4 WMMAs per tile) ----
    v16h a1k[4];
#pragma unroll
    for (int kk = 0; kk < 4; ++kk) a1k[kk] = load_a_frag(sA1, Hh1, kk * 32, lane);
#pragma unroll
    for (int n = 0; n < Hh2 / 16; ++n) {
        v8f c = {};
#pragma unroll
        for (int kk = 0; kk < 4; ++kk)
            c = wmma16(a1k[kk], load_b_frag(W2, Hh2, kk * 32, n * 16, lane), c);
        const int col  = n * 16 + ncol;
        const float bv = b2[col];
#pragma unroll
        for (int r = 0; r < 8; ++r) {
            int m   = mrow + r;
            float z = c[r] + bv;
            float sp, sg;
            act_pair(z, sp, sg);
            sA2[m * Hh2 + col] = sp;
            sS2[m * Hh2 + col] = sg;
        }
    }
    __syncthreads();

    // ---- energy per atom + dz2 = W3 * sigmoid(z2) ----
    if (lane < 16) {
        float e = b3[0];
        for (int k = 0; k < Hh2; ++k) e += sA2[lane * Hh2 + k] * W3[k];
        sE[lane] = e;
    }
    for (int i = lane; i < 16 * Hh2; i += 32) sDZ2[i] = W3[i & (Hh2 - 1)] * sS2[i];
    __syncthreads();

    // ---- bwd layer 2: da1 = dz2 @ W2^T ; dz1 = da1 * sigmoid(z1) ----
    v16h dz2k0 = load_a_frag(sDZ2, Hh2, 0,  lane);
    v16h dz2k1 = load_a_frag(sDZ2, Hh2, 32, lane);
#pragma unroll
    for (int n = 0; n < Hh1 / 16; ++n) {
        v8f c = {};
        c = wmma16(dz2k0, load_bt_frag(W2, Hh2, 0,  n * 16, lane), c);
        c = wmma16(dz2k1, load_bt_frag(W2, Hh2, 32, n * 16, lane), c);
        const int col = n * 16 + ncol;
#pragma unroll
        for (int r = 0; r < 8; ++r) {
            int m = mrow + r;
            sDZ1[m * Hh1 + col] = c[r] * sS1[m * Hh1 + col];
        }
    }
    __syncthreads();

    // ---- bwd layer 1: g = dz1 @ W1^T  -> write to workspace ----
    v16h dz1k[4];
#pragma unroll
    for (int kk = 0; kk < 4; ++kk) dz1k[kk] = load_a_frag(sDZ1, Hh1, kk * 32, lane);
    float* gout = g + (size_t)(b * Nn + m0) * Ff;
#pragma unroll
    for (int n = 0; n < Ff / 16; ++n) {
        v8f c = {};
#pragma unroll
        for (int kk = 0; kk < 4; ++kk)
            c = wmma16(dz1k[kk], load_bt_frag(W1, Hh1, kk * 32, n * 16, lane), c);
        const int col = n * 16 + ncol;
#pragma unroll
        for (int r = 0; r < 8; ++r) {
            int m = mrow + r;
            gout[m * Ff + col] = c[r];
        }
    }

    // ---- deterministic per-tile energy partial ----
    if (lane == 0) {
        float s = 0.f;
        for (int m = 0; m < 16; ++m) s += sE[m];
        partE[tile] = s;
    }
}

// ---------------------------------------------------------------------------
// Kernel C: Etot[b] = sum of 16 tile partials  (d_out[0..3])
// ---------------------------------------------------------------------------
__global__ void mlff_etot_kernel(const float* __restrict__ partE, float* __restrict__ out)
{
    int t = threadIdx.x;
    if (t < Bb) {
        float s = 0.f;
        for (int i = 0; i < 16; ++i) s += partE[t * 16 + i];
        out[t] = s;
    }
}

// ---------------------------------------------------------------------------
// Kernel B: Force[b,i,c] = sum_{j,f} g[b,j,f] * dfeat[b,i,j,f,c]
// Pure HBM stream: 201 MB of dfeat read exactly once (> 192 MB L2), so the
// dfeat loads are NON-TEMPORAL (TH=NT) to avoid thrashing L2 and keep g
// resident there. One block per (b,i) row; g[b] (64 KB) staged in dynamic
// LDS; each thread consumes groups of 3 float4 = 12 elements = 4 complete
// k values (c = idx%3 is compile-time). Deterministic LDS tree reduction.
// ---------------------------------------------------------------------------
__global__ __launch_bounds__(256) void mlff_force_kernel(
    const float* __restrict__ dfeat, const float* __restrict__ g,
    float* __restrict__ out)
{
    extern __shared__ float sg[];            // Nn*Ff = 16384 floats (64 KB)
    __shared__ float red[256 * 3];

    const int tid = threadIdx.x;
    const int bi  = blockIdx.x;              // b*N + i
    const int b   = bi >> 8;

    // stage g[b] into LDS (vectorized; regular temporal policy -> L2-resident)
    const v4f* gb4 = (const v4f*)(g + (size_t)b * (Nn * Ff));
    v4f* sg4 = (v4f*)sg;
    for (int i = tid; i < (Nn * Ff) / 4; i += 256) sg4[i] = gb4[i];
    __syncthreads();

    const v4f* row = (const v4f*)(dfeat + (size_t)bi * (Nn * Ff * 3));
    float acc0 = 0.f, acc1 = 0.f, acc2 = 0.f;

    // (N*F*3)/12 = 4096 groups of 12 contiguous floats (= 4 complete k each)
    const int ngrp = (Nn * Ff * 3) / 12;
    for (int grp = tid; grp < ngrp; grp += 256) {
        // stream-ahead hint for the next iteration of this thread
        if (grp + 256 < ngrp)
            __builtin_prefetch((const void*)&row[3 * (grp + 256)], 0, 0);

        v4f v0 = __builtin_nontemporal_load(&row[3 * grp + 0]);
        v4f v1 = __builtin_nontemporal_load(&row[3 * grp + 1]);
        v4f v2 = __builtin_nontemporal_load(&row[3 * grp + 2]);
        float vv[12] = { v0[0], v0[1], v0[2], v0[3],
                         v1[0], v1[1], v1[2], v1[3],
                         v2[0], v2[1], v2[2], v2[3] };
        const int k0 = 4 * grp;
#pragma unroll
        for (int idx = 0; idx < 12; ++idx) {
            float t = vv[idx] * sg[k0 + idx / 3];   // idx/3, idx%3 compile-time
            if      ((idx % 3) == 0) acc0 += t;
            else if ((idx % 3) == 1) acc1 += t;
            else                     acc2 += t;
        }
    }

    red[tid * 3 + 0] = acc0;
    red[tid * 3 + 1] = acc1;
    red[tid * 3 + 2] = acc2;
    __syncthreads();
#pragma unroll
    for (int s = 128; s > 0; s >>= 1) {
        if (tid < s) {
            red[tid * 3 + 0] += red[(tid + s) * 3 + 0];
            red[tid * 3 + 1] += red[(tid + s) * 3 + 1];
            red[tid * 3 + 2] += red[(tid + s) * 3 + 2];
        }
        __syncthreads();
    }
    if (tid == 0) {
        out[Bb + bi * 3 + 0] = red[0];
        out[Bb + bi * 3 + 1] = red[1];
        out[Bb + bi * 3 + 2] = red[2];
    }
}

// ---------------------------------------------------------------------------
// Launch. Inputs (setup_inputs order):
//  0 image[B,N,F] f32 | 1 dfeat[B,N,N,F,3] f32 | 2 neighbor (unused)
//  3 Egroup_weight (unused) | 4 divider (unused)
//  5 W1 | 6 b1 | 7 W2 | 8 b2 | 9 W3 | 10 b3
// Output: Etot[4] ++ Force[4*256*3]  (3076 f32)
// Workspace: g (B*N*F f32 = 256 KB) + 64 partial-energy floats.
// ---------------------------------------------------------------------------
extern "C" void kernel_launch(void* const* d_in, const int* in_sizes, int n_in,
                              void* d_out, int out_size, void* d_ws, size_t ws_size,
                              hipStream_t stream) {
    const float* image = (const float*)d_in[0];
    const float* dfeat = (const float*)d_in[1];
    const float* W1 = (const float*)d_in[5];
    const float* b1 = (const float*)d_in[6];
    const float* W2 = (const float*)d_in[7];
    const float* b2 = (const float*)d_in[8];
    const float* W3 = (const float*)d_in[9];
    const float* b3 = (const float*)d_in[10];

    float* g     = (float*)d_ws;                       // B*N*F floats
    float* partE = g + (size_t)Bb * Nn * Ff;           // 64 floats
    float* out   = (float*)d_out;

    // MLP forward+backward (WMMA), 64 tiles of 16 atoms
    mlff_mlp_grad_kernel<<<dim3(Bb * Nn / 16), dim3(32), 0, stream>>>(
        image, W1, b1, W2, b2, W3, b3, g, partE);

    // Etot reduction
    mlff_etot_kernel<<<dim3(1), dim3(32), 0, stream>>>(partE, out);

    // Force streaming einsum: one block per (b, i), 64 KB dynamic LDS for g[b]
    mlff_force_kernel<<<dim3(Bb * Nn), dim3(256), Nn * Ff * sizeof(float), stream>>>(
        dfeat, g, out);
}
